// TextCnn_81535659147319
// MI455X (gfx1250) — matrix operand order repaired and checked by reading, not compiled, not run on hardware
//
#include <hip/hip_runtime.h>
#include <hip/hip_bf16.h>

// ---------------- problem constants ----------------
#define BB   32
#define SS   4096
#define DD   256
#define TS   64            // tokens per workgroup tile
#define XROWS (TS + 4)     // token tile + 2-halo each side
#define XROW  264          // ushorts per X row (256 + 8 pad -> 4-bank rotation)
#define HROW  776          // ushorts per H row (768 + 8 pad)
#define GROW  260          // floats per G row  (256 + 4 pad)

#define X_BYTES (XROWS * XROW * 2)   // 35904
#define H_BYTES (TS * HROW * 2)      // 99328
#define LDS_TOTAL (X_BYTES + H_BYTES) // 135232 (<160KB -> 2 WG/WGP)

// conv-weight fragments: 12 taps x 16 Mtiles x 8 Ktiles x 32 lanes x 16 bf16
#define CWN (12 * 16 * 8 * 32 * 16)   // 786432 elems
// linear fragments: 16 Mtiles x 24 Ktiles x 32 lanes x 16 bf16
#define LWN (16 * 24 * 32 * 16)       // 196608 elems

typedef __attribute__((ext_vector_type(16))) __bf16 v16bf;
typedef __attribute__((ext_vector_type(8)))  float  v8f;

union Frag {
    v16bf v;
    uint4 q[2];
};

__device__ __forceinline__ unsigned short f2bf(float f) {
    unsigned int u = __float_as_uint(f);
    unsigned int r = u + 0x7fffu + ((u >> 16) & 1u);   // round-to-nearest-even
    return (unsigned short)(r >> 16);
}

// A-matrix (16x32 bf16, MxK) lane layout (contiguous k, matching B loads):
// lane = half*16 + m ; element j of v16bf
// K = half*16 + j
__device__ __forceinline__ int a_frag_k(int lane, int j) {
    return (lane >> 4) * 16 + j;
}

// ---------------- prep: permute fp32 weights -> bf16 WMMA A-fragments ----------------
__global__ void textcnn_prep_kernel(const float* __restrict__ w3,
                                    const float* __restrict__ w4,
                                    const float* __restrict__ w5,
                                    const float* __restrict__ lin_w,
                                    unsigned short* __restrict__ cw,
                                    unsigned short* __restrict__ lw) {
    int idx = blockIdx.x * blockDim.x + threadIdx.x;
    if (idx < CWN) {
        // layout: (((tap*16 + mt)*8 + kt)*32 + lane)*16 + j
        int j = idx & 15;
        int r = idx >> 4;
        int lane = r & 31; r >>= 5;
        int kt = r & 7;    r >>= 3;
        int mt = r & 15;
        int tap = r >> 4;
        int m = lane & 15;
        int K = a_frag_k(lane, j);
        int c_out = mt * 16 + m;
        int c_in  = kt * 32 + K;
        const float* w; int k, jj;
        if (tap < 3)      { w = w3; k = 3; jj = tap; }
        else if (tap < 7) { w = w4; k = 4; jj = tap - 3; }
        else              { w = w5; k = 5; jj = tap - 7; }
        cw[idx] = f2bf(w[(c_out * DD + c_in) * k + jj]);
    } else if (idx < CWN + LWN) {
        int idx2 = idx - CWN;
        // layout: ((mt*24 + kt)*32 + lane)*16 + j
        int j = idx2 & 15;
        int r = idx2 >> 4;
        int lane = r & 31; r >>= 5;
        int kt = r % 24;
        int mt = r / 24;
        int m = lane & 15;
        int K = a_frag_k(lane, j);
        int row = mt * 16 + m;
        int col = kt * 32 + K;     // < 768
        lw[idx2] = f2bf(lin_w[row * (3 * DD) + col]);
    }
}

// ---------------- fused conv(3,4,5) + relu + linear + LN + residual ----------------
__global__ __launch_bounds__(256, 2)
void textcnn_main_kernel(const float* __restrict__ emb,
                         const unsigned char* __restrict__ mask,
                         const float* __restrict__ b3,
                         const float* __restrict__ b4,
                         const float* __restrict__ b5,
                         const float* __restrict__ lin_b,
                         const unsigned short* __restrict__ cw,
                         const unsigned short* __restrict__ lw,
                         float* __restrict__ out) {
    __shared__ __align__(16) unsigned char smem[LDS_TOTAL];
    unsigned short* X = (unsigned short*)(smem);
    unsigned short* H = (unsigned short*)(smem + X_BYTES);
    float*          G = (float*)(smem + X_BYTES);   // aliases H (barrier-separated)

    const int tile = blockIdx.x;     // 0..63
    const int b    = blockIdx.y;     // 0..31
    const int t0   = tile * TS;
    const int tid  = threadIdx.x;

    const float* embB = emb + (size_t)b * SS * DD;

    // ---- stage 1: load emb tile (+halo) as bf16 into LDS ----
    for (int g = tid; g < XROWS * (DD / 4); g += 256) {
        int row = g >> 6;                 // 0..67
        int c4  = (g & 63) * 4;           // channel base
        int s   = t0 - 2 + row;
        float4 v = make_float4(0.f, 0.f, 0.f, 0.f);
        if (s >= 0 && s < SS) v = *(const float4*)(embB + (size_t)s * DD + c4);
        uint2 p;
        p.x = (unsigned)f2bf(v.x) | ((unsigned)f2bf(v.y) << 16);
        p.y = (unsigned)f2bf(v.z) | ((unsigned)f2bf(v.w) << 16);
        *(uint2*)(X + row * XROW + c4) = p;
    }
    __syncthreads();

    const int wave    = tid >> 5;
    const int lane    = tid & 31;
    const int half    = lane >> 4;
    const int tokLane = lane & 15;
    const int nt      = wave & 3;      // N-tile (16 tokens)
    const int g2      = wave >> 2;     // M-group (0/1)
    const int tok     = nt * 16 + tokLane;

    // ---- stage 2: conv GEMM, H(768 x 64) = sum_taps W_tap * X_shift ----
    for (int kb = 0; kb < 3; ++kb) {
        v8f acc[8];
        #pragma unroll
        for (int i = 0; i < 8; ++i) acc[i] = (v8f){0.f,0.f,0.f,0.f,0.f,0.f,0.f,0.f};

        const int ntaps   = 3 + kb;
        const int tapBase = (kb == 0) ? 0 : ((kb == 1) ? 3 : 7);
        const int ctr     = (kb == 0) ? 1 : 2;

        for (int t = 0; t < ntaps; ++t) {
            const int off    = t - ctr;                       // token shift
            const int tapIdx = tapBase + t;
            const int xrow   = tok + 2 + off;                 // 0..67
            #pragma unroll
            for (int kt = 0; kt < 8; ++kt) {
                Frag bf;
                const uint4* bp = (const uint4*)(X + xrow * XROW + kt * 32 + half * 16);
                bf.q[0] = bp[0]; bf.q[1] = bp[1];
                #pragma unroll
                for (int i = 0; i < 8; ++i) {
                    const int mtk = g2 * 8 + i;               // Mtile within k-block
                    Frag af;
                    const uint4* ap = (const uint4*)(cw +
                        (((size_t)(tapIdx * 16 + mtk) * 8 + kt) * 32 + lane) * 16);
                    af.q[0] = ap[0]; af.q[1] = ap[1];
                    acc[i] = __builtin_amdgcn_wmma_f32_16x16x32_bf16(
                        false, af.v, false, bf.v, (short)0, acc[i], false, false);
                }
            }
        }
        // epilogue: +bias, ReLU, pack bf16 -> LDS H
        const float* bias = (kb == 0) ? b3 : ((kb == 1) ? b4 : b5);
        #pragma unroll
        for (int i = 0; i < 8; ++i) {
            const int mtk = g2 * 8 + i;
            const int cl  = mtk * 16 + half * 8;     // c_out base within k-block
            const int c   = kb * 256 + cl;           // H channel base
            unsigned int p[4];
            #pragma unroll
            for (int r = 0; r < 4; ++r) {
                float v0 = acc[i][2 * r]     + bias[cl + 2 * r];
                float v1 = acc[i][2 * r + 1] + bias[cl + 2 * r + 1];
                v0 = v0 > 0.f ? v0 : 0.f;
                v1 = v1 > 0.f ? v1 : 0.f;
                p[r] = (unsigned)f2bf(v0) | ((unsigned)f2bf(v1) << 16);
            }
            *(uint4*)(H + tok * HROW + c) = make_uint4(p[0], p[1], p[2], p[3]);
        }
    }
    __syncthreads();

    // ---- stage 3: projection GEMM, G(256 x 64) = LinW(256x768) * relu(H) ----
    v8f acc2[8];
    #pragma unroll
    for (int i = 0; i < 8; ++i) acc2[i] = (v8f){0.f,0.f,0.f,0.f,0.f,0.f,0.f,0.f};

    for (int kt = 0; kt < 24; ++kt) {
        Frag bf;
        const uint4* bp = (const uint4*)(H + tok * HROW + kt * 32 + half * 16);
        bf.q[0] = bp[0]; bf.q[1] = bp[1];
        #pragma unroll
        for (int i = 0; i < 8; ++i) {
            const int mt = g2 * 8 + i;
            Frag af;
            const uint4* ap = (const uint4*)(lw + ((size_t)(mt * 24 + kt) * 32 + lane) * 16);
            af.q[0] = ap[0]; af.q[1] = ap[1];
            acc2[i] = __builtin_amdgcn_wmma_f32_16x16x32_bf16(
                false, af.v, false, bf.v, (short)0, acc2[i], false, false);
        }
    }
    __syncthreads();   // all H reads done; safe to overwrite aliased region with G

    #pragma unroll
    for (int i = 0; i < 8; ++i) {
        const int c = (g2 * 8 + i) * 16 + half * 8;
        float4 lo, hi;
        lo.x = acc2[i][0] + lin_b[c + 0];
        lo.y = acc2[i][1] + lin_b[c + 1];
        lo.z = acc2[i][2] + lin_b[c + 2];
        lo.w = acc2[i][3] + lin_b[c + 3];
        hi.x = acc2[i][4] + lin_b[c + 4];
        hi.y = acc2[i][5] + lin_b[c + 5];
        hi.z = acc2[i][6] + lin_b[c + 6];
        hi.w = acc2[i][7] + lin_b[c + 7];
        *(float4*)(G + tok * GROW + c)     = lo;
        *(float4*)(G + tok * GROW + c + 4) = hi;
    }
    __syncthreads();

    // ---- stage 4: LayerNorm + residual + mask, fp32 out ----
    const int tk = tid >> 2;                 // token in tile (0..63)
    const int q  = tid & 3;                  // channel quarter
    const float* rowp = G + tk * GROW + q * 64;
    float s = 0.f, ss = 0.f;
    #pragma unroll
    for (int j = 0; j < 64; j += 4) {
        float4 v = *(const float4*)(rowp + j);
        s  += v.x + v.y + v.z + v.w;
        ss += v.x * v.x + v.y * v.y + v.z * v.z + v.w * v.w;
    }
    s  += __shfl_xor(s, 1);  s  += __shfl_xor(s, 2);
    ss += __shfl_xor(ss, 1); ss += __shfl_xor(ss, 2);
    const float mu  = s * (1.f / 256.f);
    const float var = ss * (1.f / 256.f) - mu * mu;
    const float rs  = rsqrtf(var + 1e-5f);

    const int sGlob = t0 + tk;
    const bool mk = mask[(size_t)b * SS + sGlob] != 0;
    const float* e = embB + (size_t)sGlob * DD + q * 64;
    float* o = out + ((size_t)b * SS + sGlob) * (size_t)DD + q * 64;
    #pragma unroll
    for (int j = 0; j < 64; j += 4) {
        float4 gv = *(const float4*)(rowp + j);
        float4 ev = *(const float4*)(e + j);
        float4 ov;
        ov.x = mk ? ev.x + (gv.x - mu) * rs : 0.f;
        ov.y = mk ? ev.y + (gv.y - mu) * rs : 0.f;
        ov.z = mk ? ev.z + (gv.z - mu) * rs : 0.f;
        ov.w = mk ? ev.w + (gv.w - mu) * rs : 0.f;
        *(float4*)(o + j) = ov;
    }
}

extern "C" void kernel_launch(void* const* d_in, const int* in_sizes, int n_in,
                              void* d_out, int out_size, void* d_ws, size_t ws_size,
                              hipStream_t stream) {
    const float*         emb   = (const float*)d_in[0];
    const unsigned char* mask  = (const unsigned char*)d_in[1];
    const float*         w3    = (const float*)d_in[2];
    const float*         b3    = (const float*)d_in[3];
    const float*         w4    = (const float*)d_in[4];
    const float*         b4    = (const float*)d_in[5];
    const float*         w5    = (const float*)d_in[6];
    const float*         b5    = (const float*)d_in[7];
    const float*         lin_w = (const float*)d_in[8];
    const float*         lin_b = (const float*)d_in[9];
    float*               out   = (float*)d_out;

    unsigned short* cw = (unsigned short*)d_ws;
    unsigned short* lw = cw + CWN;

    // stage 0: permute weights into bf16 WMMA fragments (deterministic, every call)
    const int prepThreads = CWN + LWN;
    textcnn_prep_kernel<<<(prepThreads + 255) / 256, 256, 0, stream>>>(
        w3, w4, w5, lin_w, cw, lw);

    // fused main kernel: one workgroup per (64-token tile, batch row)
    dim3 grid(SS / TS, BB);
    textcnn_main_kernel<<<grid, 256, 0, stream>>>(
        emb, mask, b3, b4, b5, lin_b, cw, lw, out);
}
